// TBCNNClassifier_3899830305139
// MI455X (gfx1250) — matrix operand order repaired and checked by reading, not compiled, not run on hardware
//
#include <hip/hip_runtime.h>

#define NNODES 409600   // G * PER
#define NEDGES 409088   // G * (PER-1)
#define NGRAPH 512
#define PER    800
#define HDIM   256
#define KDIM   768      // 3 * HDIM stacked [left | right | h]
#define NCLS   104

#define KT_TOTAL 24     // K tiles of 32
#define KT_PASS  6      // K tiles staged per LDS pass
#define NPASS    4
#define WAVES_PER_BLOCK 8
#define CHUNK_SHORTS (KT_PASS * 16 * 32 * 16)   // 49152 bf16 = 98304 bytes per pass

typedef __attribute__((ext_vector_type(16))) __bf16 v16bf;
typedef __attribute__((ext_vector_type(8)))  float  v8f;
typedef __attribute__((ext_vector_type(8)))  unsigned int v8u;

__device__ __forceinline__ float bf2f(unsigned short u) {
    unsigned int x = ((unsigned int)u) << 16;
    float f; __builtin_memcpy(&f, &x, 4); return f;
}
__device__ __forceinline__ unsigned short f2bf(float f) {
    unsigned int u; __builtin_memcpy(&u, &f, 4);
    u += 0x7FFFu + ((u >> 16) & 1u);   // round-to-nearest-even
    return (unsigned short)(u >> 16);
}

__device__ __forceinline__ void async_g2l_b128(unsigned lds_byte_addr, const void* gaddr) {
    // GLOBAL_LOAD_ASYNC_TO_LDS_B128 (GV mode): VDST = LDS byte address, VADDR = 64-bit global
    asm volatile("global_load_async_to_lds_b128 %0, %1, off"
                 :: "v"(lds_byte_addr), "v"(gaddr) : "memory");
}
__device__ __forceinline__ void wait_asynccnt0() {
#if __has_builtin(__builtin_amdgcn_s_wait_asynccnt)
    __builtin_amdgcn_s_wait_asynccnt(0);
#else
    asm volatile("s_wait_asynccnt 0x0" ::: "memory");
#endif
}

// ---- Pack B = [W_left; W_right; W_top] (768x256 f32) into WMMA bf16 fragment order.
// Linear layout: ((kt*16 + ntile)*32 + lane)*16 + t
//   lane<16  -> K=kt*32+t,    col=ntile*16+lane
//   lane>=16 -> K=kt*32+16+t, col=ntile*16+lane-16
__global__ void k_pack(const float* __restrict__ wl, const float* __restrict__ wr,
                       const float* __restrict__ wt, unsigned short* __restrict__ Bf) {
    int i = blockIdx.x * 256 + threadIdx.x;            // 768*256 = 196608 total
    int kt = i / 8192;   int r  = i % 8192;
    int ntile = r / 512; int r2 = r % 512;
    int lane = r2 / 16;  int t  = r2 % 16;
    int k   = kt * 32 + ((lane >= 16) ? 16 : 0) + t;
    int col = ntile * 16 + (lane & 15);
    float v = (k < 256) ? wl[k * HDIM + col]
            : (k < 512) ? wr[(k - 256) * HDIM + col]
                        : wt[(k - 512) * HDIM + col];
    Bf[i] = f2bf(v);
}

// ---- Embedding gather -> A[:,512:768] (bf16), zero A[:,0:512] (left/right slots)
__global__ void k_embed(const int* __restrict__ node_type, const float* __restrict__ emb,
                        unsigned short* __restrict__ A) {
    int idx = blockIdx.x * 256 + threadIdx.x;          // NNODES*256 threads
    int n = idx >> 8; int c = idx & 255;
    unsigned short* row = A + (long)n * KDIM;
    row[c] = 0; row[256 + c] = 0;
    row[512 + c] = f2bf(emb[(long)node_type[n] * HDIM + c]);
}

// ---- Edges sorted by dst: mark first edge per dst
__global__ void k_starts(const int* __restrict__ dst, int* __restrict__ starts, int E) {
    int e = blockIdx.x * 256 + threadIdx.x;
    if (e >= E) return;
    int d = dst[e];
    if (e == 0 || dst[e - 1] != d) starts[d] = e;
}

// ---- Per-parent aggregation: left = sum (1-a) h_child, right = sum a h_child (one wave / node)
__global__ void k_agg(const int* __restrict__ esrc, const float* __restrict__ alpha,
                      const int* __restrict__ cc, const int* __restrict__ starts,
                      unsigned short* __restrict__ A) {
    int n    = (blockIdx.x * blockDim.x + threadIdx.x) >> 5;
    int lane = threadIdx.x & 31;
    int c    = cc[n];
    if (c == 0) return;
    int s  = starts[n];
    int c0 = lane * 8;
    float L[8], R[8];
    #pragma unroll
    for (int j = 0; j < 8; j++) { L[j] = 0.f; R[j] = 0.f; }
    for (int e = s; e < s + c; e++) {
        int sn = esrc[e]; float a = alpha[e]; float oa = 1.0f - a;
        uint4 q = *(const uint4*)(A + (long)sn * KDIM + 512 + c0);
        unsigned short hs[8]; __builtin_memcpy(hs, &q, 16);
        #pragma unroll
        for (int j = 0; j < 8; j++) { float f = bf2f(hs[j]); L[j] += oa * f; R[j] += a * f; }
    }
    unsigned short ol[8], orr[8];
    #pragma unroll
    for (int j = 0; j < 8; j++) { ol[j] = f2bf(L[j]); orr[j] = f2bf(R[j]); }
    uint4 ql, qr; __builtin_memcpy(&ql, ol, 16); __builtin_memcpy(&qr, orr, 16);
    *(uint4*)(A + (long)n * KDIM + c0)       = ql;
    *(uint4*)(A + (long)n * KDIM + 256 + c0) = qr;
}

// ---- Fused GEMM: hf = relu(A[N,768] @ B[768,256] + bias); where(cc==0) keep embedding.
// 8 waves/block, one 16-row M-tile per wave, full 256 output cols in-register (16 v8f acc).
// B fragments double-buffered in LDS via async global->LDS copies (ASYNCcnt).
__global__ void __launch_bounds__(256) k_gemm(const unsigned short* __restrict__ A,
                                              const unsigned short* __restrict__ Bf,
                                              const float* __restrict__ bias,
                                              const int* __restrict__ cc,
                                              float* __restrict__ hf) {
    __shared__ unsigned short Bsh[2][CHUNK_SHORTS];    // 2 x 96 KB

    const int tid  = threadIdx.x;
    const int lane = tid & 31;
    const int wid  = tid >> 5;                          // wave id in block
    const int m0   = (blockIdx.x * WAVES_PER_BLOCK + wid) * 16;
    const int rsel = lane & 15;
    const int hsel = lane >> 4;

    // ---- stage pass 0 into buffer 0 (async), then wait + barrier
    {
        const unsigned short* src = Bf;                 // pass 0 chunk base
        unsigned lbase = (unsigned)(uintptr_t)(&Bsh[0][0]);
        for (int i = 0; i < CHUNK_SHORTS / (8 * 256); i++) {   // 24 units of 16B per thread
            int u = i * 256 + tid;
            async_g2l_b128(lbase + u * 16, src + u * 8);
        }
        wait_asynccnt0();
        __syncthreads();
    }

    v8f acc[16];
    #pragma unroll
    for (int t = 0; t < 16; t++) {
        #pragma unroll
        for (int j = 0; j < 8; j++) acc[t][j] = 0.0f;
    }
    const long rowA = (long)(m0 + rsel) * KDIM;

    for (int p = 0; p < NPASS; p++) {
        // ---- prefetch next pass into the other buffer (async, overlapped with compute)
        if (p + 1 < NPASS) {
            const unsigned short* src = Bf + (long)(p + 1) * CHUNK_SHORTS;
            unsigned lbase = (unsigned)(uintptr_t)(&Bsh[(p + 1) & 1][0]);
            for (int i = 0; i < CHUNK_SHORTS / (8 * 256); i++) {
                int u = i * 256 + tid;
                async_g2l_b128(lbase + u * 16, src + u * 8);
            }
        }

        // ---- compute on current buffer
        const unsigned short* lb = &Bsh[p & 1][0];
        for (int ktl = 0; ktl < KT_PASS; ktl++) {
            int kbase = (p * KT_PASS + ktl) * 32 + hsel * 16;
            uint4 lo = *(const uint4*)(A + rowA + kbase);
            uint4 hi = *(const uint4*)(A + rowA + kbase + 8);
            union { v8u u; v16bf v; } ua;
            ua.u = (v8u){lo.x, lo.y, lo.z, lo.w, hi.x, hi.y, hi.z, hi.w};
            #pragma unroll
            for (int ntv = 0; ntv < 16; ntv++) {
                int off = ((ktl * 16 + ntv) * 32 + lane) * 16;
                uint4 b0 = *(const uint4*)(lb + off);
                uint4 b1 = *(const uint4*)(lb + off + 8);
                union { v8u u; v16bf v; } ub;
                ub.u = (v8u){b0.x, b0.y, b0.z, b0.w, b1.x, b1.y, b1.z, b1.w};
                acc[ntv] = __builtin_amdgcn_wmma_f32_16x16x32_bf16(
                    false, ua.v, false, ub.v, (short)0, acc[ntv], false, false);
            }
        }

        if (p + 1 < NPASS) {
            wait_asynccnt0();
            __syncthreads();
        }
    }

    // ---- Epilogue: bias + relu + where(child_count>0, conv, h_emb)
    int ccv[8];
    #pragma unroll
    for (int r = 0; r < 8; r++) ccv[r] = cc[m0 + hsel * 8 + r];
    #pragma unroll
    for (int ntv = 0; ntv < 16; ntv++) {
        int col = ntv * 16 + rsel;
        float bb = bias[col];
        #pragma unroll
        for (int r = 0; r < 8; r++) {
            int row = m0 + hsel * 8 + r;
            float v = acc[ntv][r] + bb;
            v = v > 0.0f ? v : 0.0f;
            if (ccv[r] == 0) v = bf2f(A[(long)row * KDIM + 512 + col]);
            hf[(long)row * HDIM + col] = v;
        }
    }
}

// ---- Gate score: gate[n] = hf[n,:] . gate_w + gate_b (one wave per node)
__global__ void k_gate(const float* __restrict__ hf, const float* __restrict__ gw,
                       const float* __restrict__ gb, float* __restrict__ gate) {
    int n    = (blockIdx.x * blockDim.x + threadIdx.x) >> 5;
    int lane = threadIdx.x & 31;
    const float4 a  = *(const float4*)(hf + (long)n * HDIM + lane * 8);
    const float4 b  = *(const float4*)(hf + (long)n * HDIM + lane * 8 + 4);
    const float4 wa = *(const float4*)(gw + lane * 8);
    const float4 wb = *(const float4*)(gw + lane * 8 + 4);
    float d = a.x * wa.x + a.y * wa.y + a.z * wa.z + a.w * wa.w
            + b.x * wb.x + b.y * wb.y + b.z * wb.z + b.w * wb.w;
    #pragma unroll
    for (int o = 16; o > 0; o >>= 1) d += __shfl_xor(d, o, 32);
    if (lane == 0) gate[n] = d + gb[0];
}

// ---- Per-graph softmax attention pooling (graphs are contiguous 800-node runs)
__global__ void k_pool(const float* __restrict__ gate, const float* __restrict__ hf,
                       float* __restrict__ pooled) {
    __shared__ float red[256];
    __shared__ float ew[PER];
    int g = blockIdx.x, tid = threadIdx.x;
    const float* gg = gate + (long)g * PER;
    float m = -3.4e38f;
    for (int n = tid; n < PER; n += 256) m = fmaxf(m, gg[n]);
    red[tid] = m; __syncthreads();
    for (int s = 128; s > 0; s >>= 1) { if (tid < s) red[tid] = fmaxf(red[tid], red[tid + s]); __syncthreads(); }
    float gmax = red[0]; __syncthreads();
    float sum = 0.f;
    for (int n = tid; n < PER; n += 256) { float e = __expf(gg[n] - gmax); ew[n] = e; sum += e; }
    red[tid] = sum; __syncthreads();
    for (int s = 128; s > 0; s >>= 1) { if (tid < s) red[tid] += red[tid + s]; __syncthreads(); }
    float inv = 1.0f / red[0];
    float acc = 0.f;
    const float* hb = hf + (long)g * PER * HDIM + tid;   // tid = channel
    for (int n = 0; n < PER; n++) acc += ew[n] * hb[(long)n * HDIM];
    pooled[(long)g * HDIM + tid] = acc * inv;
}

// ---- Classifier: out[g,c] = pooled[g,:] . cls_w[:,c] + cls_b[c]
__global__ void k_cls(const float* __restrict__ pooled, const float* __restrict__ cw,
                      const float* __restrict__ cb, float* __restrict__ out) {
    int g = blockIdx.x, c = threadIdx.x;                 // 512 x 104
    float acc = cb[c];
    for (int k = 0; k < HDIM; k++) acc += pooled[(long)g * HDIM + k] * cw[k * NCLS + c];
    out[(long)g * NCLS + c] = acc;
}

extern "C" void kernel_launch(void* const* d_in, const int* in_sizes, int n_in,
                              void* d_out, int out_size, void* d_ws, size_t ws_size,
                              hipStream_t stream) {
    const int*   node_type = (const int*)  d_in[0];
    const int*   edge_src  = (const int*)  d_in[1];
    const int*   edge_dst  = (const int*)  d_in[2];
    const float* alpha     = (const float*)d_in[3];
    const int*   ccnt      = (const int*)  d_in[4];
    // d_in[5] = graph_ids (implicit: contiguous 800-node runs)
    const float* emb       = (const float*)d_in[6];
    const float* W_left    = (const float*)d_in[7];
    const float* W_right   = (const float*)d_in[8];
    const float* W_top     = (const float*)d_in[9];
    const float* b_conv    = (const float*)d_in[10];
    const float* gate_w    = (const float*)d_in[11];
    const float* gate_b    = (const float*)d_in[12];
    const float* cls_w     = (const float*)d_in[13];
    const float* cls_b     = (const float*)d_in[14];
    float* out = (float*)d_out;

    char* ws = (char*)d_ws;
    const size_t A_BYTES  = (size_t)NNODES * KDIM * 2;          // 629,145,600
    const size_t HF_BYTES = (size_t)NNODES * HDIM * 4;          // 419,430,400
    const size_t BF_BYTES = (size_t)KDIM * HDIM * 2;            // 393,216
    const size_t ST_BYTES = (size_t)NNODES * 4;
    const size_t GT_BYTES = (size_t)NNODES * 4;
    unsigned short* A      = (unsigned short*)ws;
    float*          hf     = (float*)(ws + A_BYTES);
    unsigned short* Bf     = (unsigned short*)(ws + A_BYTES + HF_BYTES);
    int*            starts = (int*)(ws + A_BYTES + HF_BYTES + BF_BYTES);
    float*          gate   = (float*)(ws + A_BYTES + HF_BYTES + BF_BYTES + ST_BYTES);
    float*          pooled = (float*)(ws + A_BYTES + HF_BYTES + BF_BYTES + ST_BYTES + GT_BYTES);

    k_pack  <<<(KDIM * HDIM) / 256, 256, 0, stream>>>(W_left, W_right, W_top, Bf);
    k_embed <<<NNODES, 256, 0, stream>>>(node_type, emb, A);
    k_starts<<<(NEDGES + 255) / 256, 256, 0, stream>>>(edge_dst, starts, NEDGES);
    k_agg   <<<NNODES / 8, 256, 0, stream>>>(edge_src, alpha, ccnt, starts, A);
    k_gemm  <<<NNODES / (16 * WAVES_PER_BLOCK), 256, 0, stream>>>(A, Bf, b_conv, ccnt, hf);
    k_gate  <<<NNODES / 8, 256, 0, stream>>>(hf, gate_w, gate_b, gate);
    k_pool  <<<NGRAPH, 256, 0, stream>>>(gate, hf, pooled);
    k_cls   <<<NGRAPH, NCLS, 0, stream>>>(pooled, cls_w, cls_b, out);
}
